// GraphAttentionLayer_45689862094922
// MI455X (gfx1250) — matrix operand order repaired and checked by reading, not compiled, not run on hardware
//
#include <hip/hip_runtime.h>

// GAT layer for MI455X (gfx1250, wave32, WMMA).
// N=4096, R=4, Fin=256, Fout=64. Memory-bound on edge_attr (256MB):
// two passes over it + 256MB e writeback ~= 770MB => ~33us at 23.3TB/s.
// Contraction runs on v_wmma_f32_16x16x32_f16 (f16 A/B, f32 accum).

#define NN    4096
#define FIN   256
#define FOUT  64
#define RDIM  4
#define ALPHA 0.2f
#define NEGINF (-9.0e15f)

typedef _Float16 v16h __attribute__((ext_vector_type(16)));
typedef _Float16 v8h  __attribute__((ext_vector_type(8)));
typedef float    v8f  __attribute__((ext_vector_type(8)));
typedef float    v4f  __attribute__((ext_vector_type(4)));

union AFrag { v16h v; v8h h[2]; };

static __device__ __forceinline__ v8f wmma_f16(v16h a, v16h b, v8f c) {
  // D = A(16x32 f16) * B(32x16 f16) + C(16x16 f32)
  return __builtin_amdgcn_wmma_f32_16x16x32_f16(false, a, false, b,
                                                (short)0, c, false, false);
}

static __device__ __forceinline__ v8f zero_v8f() {
  v8f z;
  #pragma unroll
  for (int e = 0; e < 8; ++e) z[e] = 0.0f;
  return z;
}

// ---------------------------------------------------------------------------
// Kernel 1: Wh = h @ W   (4096x256 * 256x64), f16 WMMA, f32 accum.
// Also writes WhT (f16, [FOUT][N]) so pass-4 B-fragments are contiguous.
// Block = 128 threads (4 waves); each wave owns a 16x64 tile of Wh.
// ---------------------------------------------------------------------------
__global__ __launch_bounds__(128)
void gat_wh_gemm(const float* __restrict__ h, const float* __restrict__ W,
                 float* __restrict__ Wh, _Float16* __restrict__ WhT) {
  // W transposed to f16 in LDS: wt[f][k], padded row stride (16B aligned).
  __shared__ __align__(16) _Float16 wt[FOUT][FIN + 8];
  const int tid = threadIdx.x;
  for (int idx = tid; idx < FIN * FOUT; idx += 128) {
    // W row-major [k][f]
    wt[idx & (FOUT - 1)][idx >> 6] = (_Float16)W[idx];
  }
  __syncthreads();

  const int wv = tid >> 5;
  const int lane = tid & 31;
  const int hi = lane >> 4;      // lane half selects K sub-ranges per ISA layout
  const int mrow = lane & 15;    // A-matrix row M for this lane
  const int irow = blockIdx.x * 64 + wv * 16 + mrow;

  v8f acc[4];
  #pragma unroll
  for (int n = 0; n < 4; ++n) acc[n] = zero_v8f();

  for (int k0 = 0; k0 < FIN; k0 += 32) {
    // A fragment: lanes 0-15 hold K{0..7,16..23}, lanes 16-31 K{8..15,24..31}
    float tf[16];
    const float* hp = h + (size_t)irow * FIN + k0 + hi * 8;
    *(v4f*)&tf[0]  = *(const v4f*)(hp + 0);
    *(v4f*)&tf[4]  = *(const v4f*)(hp + 4);
    *(v4f*)&tf[8]  = *(const v4f*)(hp + 16);
    *(v4f*)&tf[12] = *(const v4f*)(hp + 20);
    AFrag a;
    #pragma unroll
    for (int c = 0; c < 16; ++c) a.v[c] = (_Float16)tf[c];

    // B fragment: lane holds column N=mrow, K = k0 + {hi*8..+7, 16+hi*8..+7}
    const int kb = k0 + hi * 8;
    #pragma unroll
    for (int n = 0; n < 4; ++n) {
      const int col = n * 16 + mrow;
      AFrag b;
      b.h[0] = *(const v8h*)&wt[col][kb];
      b.h[1] = *(const v8h*)&wt[col][kb + 16];
      acc[n] = wmma_f16(a.v, b.v, acc[n]);
    }
  }

  // C/D layout: element v -> row M = v + hi*8, column N = mrow.
  #pragma unroll
  for (int n = 0; n < 4; ++n) {
    const int col = n * 16 + mrow;
    #pragma unroll
    for (int v = 0; v < 8; ++v) {
      const int row = blockIdx.x * 64 + wv * 16 + hi * 8 + v;
      const float cv = acc[n][v];
      Wh[(size_t)row * FOUT + col] = cv;
      WhT[(size_t)col * NN + row] = (_Float16)cv;
    }
  }
}

// ---------------------------------------------------------------------------
// Kernel 2: Wh1[i] = Wh[i,:] . a[:64], Wh2[i] = Wh[i,:] . a[64:]
// ---------------------------------------------------------------------------
__global__ __launch_bounds__(256)
void gat_attn_vec(const float* __restrict__ Wh, const float* __restrict__ a,
                  float* __restrict__ wh1, float* __restrict__ wh2) {
  const int i = blockIdx.x * 256 + threadIdx.x;
  const float* row = Wh + (size_t)i * FOUT;
  float s1 = 0.f, s2 = 0.f;
  #pragma unroll 8
  for (int f = 0; f < FOUT; ++f) {
    const float v = row[f];
    s1 += v * a[f];
    s2 += v * a[FOUT + f];
  }
  wh1[i] = s1;
  wh2[i] = s2;
}

// ---------------------------------------------------------------------------
// Kernel 3: column softmax stats (softmax over axis=1, i.e. over rows i).
// One thread per column j; online (rescaled) max/sum; coalesced over j.
// x[r,i,j] = edge>0 ? leakyrelu(wh1[i]+wh2[j]) * edge : -9e15
// ---------------------------------------------------------------------------
__global__ __launch_bounds__(256)
void gat_col_stats(const float* __restrict__ edge, const float* __restrict__ wh1,
                   const float* __restrict__ wh2, float* __restrict__ mcol,
                   float* __restrict__ sinv) {
  __shared__ float w1s[256];
  const int r = blockIdx.y;
  const int j = blockIdx.x * 256 + threadIdx.x;
  const float* col = edge + (size_t)r * NN * NN + j;
  const float w2 = wh2[j];

  float m = -3.0e38f, s = 0.0f;
  for (int ib = 0; ib < NN; ib += 256) {
    __syncthreads();
    w1s[threadIdx.x] = wh1[ib + threadIdx.x];
    __syncthreads();
    #pragma unroll 4
    for (int ii = 0; ii < 256; ++ii) {
      const float ea = col[(size_t)(ib + ii) * NN];
      float pre = w1s[ii] + w2;
      pre = (pre >= 0.f) ? pre : ALPHA * pre;
      const float x = (ea > 0.f) ? pre * ea : NEGINF;
      const float mn = fmaxf(m, x);
      s = s * __expf(m - mn) + __expf(x - mn);
      m = mn;
    }
  }
  mcol[r * NN + j] = m;
  sinv[r * NN + j] = 1.0f / s;
}

// ---------------------------------------------------------------------------
// Kernel 4: fused normalize + e-writeback + (e @ Wh) WMMA + ELU.
// Block = 128 threads (4 waves), owns 64 rows i for one r; sweeps all j in
// K=32 tiles. e tile staged in LDS as f16 -> WMMA A fragments via ds_load_b128.
// ---------------------------------------------------------------------------
__global__ __launch_bounds__(128)
void gat_apply(const float* __restrict__ edge, const float* __restrict__ wh1,
               const float* __restrict__ wh2, const float* __restrict__ mcol,
               const float* __restrict__ sinv, const _Float16* __restrict__ WhT,
               float* __restrict__ out0, float* __restrict__ out_e) {
  __shared__ __align__(16) _Float16 etile[64][40];  // 32 cols + pad, 16B-aligned rows
  __shared__ float w2s[32], ms[32], ss[32];

  const int r = blockIdx.y;
  const int i0 = blockIdx.x * 64;
  const int tid = threadIdx.x;
  const int wv = tid >> 5;
  const int lane = tid & 31;
  const int hi = lane >> 4;
  const int mrow = lane & 15;

  // stage-1 mapping: thread -> (row 0..63, 16-col half-tile)
  const int s_row = tid >> 1;
  const int s_col = (tid & 1) * 16;
  const int i_s = i0 + s_row;
  const float w1 = wh1[i_s];
  const float* erow = edge + ((size_t)r * NN + i_s) * NN;
  float* eorow = out_e + ((size_t)r * NN + i_s) * NN;

  v8f acc[4];
  #pragma unroll
  for (int n = 0; n < 4; ++n) acc[n] = zero_v8f();

  for (int j0 = 0; j0 < NN; j0 += 32) {
    if (tid < 32) {
      const int j = j0 + tid;
      w2s[tid] = wh2[j];
      ms[tid]  = mcol[r * NN + j];
      ss[tid]  = sinv[r * NN + j];
    }
    __syncthreads();

    // ---- stage 1: load edge tile, normalize, write e (f32), stash f16 ----
    {
      const float* p = erow + j0 + s_col;
      __builtin_prefetch(p + 32, 0, 1);  // next K-tile -> global_prefetch_b8
      float tf[16];
      *(v4f*)&tf[0]  = *(const v4f*)(p + 0);
      *(v4f*)&tf[4]  = *(const v4f*)(p + 4);
      *(v4f*)&tf[8]  = *(const v4f*)(p + 8);
      *(v4f*)&tf[12] = *(const v4f*)(p + 12);
      float eb[16];
      #pragma unroll
      for (int c = 0; c < 16; ++c) {
        const int jc = s_col + c;
        const float ea = tf[c];
        float pre = w1 + w2s[jc];
        pre = (pre >= 0.f) ? pre : ALPHA * pre;
        const float x = (ea > 0.f) ? pre * ea : NEGINF;
        const float ev = __expf(x - ms[jc]) * ss[jc];  // masked -> exp(-huge)=0
        eb[c] = ev;
        etile[s_row][jc] = (_Float16)ev;
      }
      float* q = eorow + j0 + s_col;
      *(v4f*)(q + 0)  = *(const v4f*)&eb[0];
      *(v4f*)(q + 4)  = *(const v4f*)&eb[4];
      *(v4f*)(q + 8)  = *(const v4f*)&eb[8];
      *(v4f*)(q + 12) = *(const v4f*)&eb[12];
    }
    __syncthreads();

    // ---- stage 2: WMMA accumulate h_prime tile (16 rows x 64 cols / wave) ----
    const int arow = wv * 16 + mrow;
    const int ka = hi ? 16 : 0;  // A layout: {0..15} / {16..31}
    AFrag a;
    a.h[0] = *(const v8h*)&etile[arow][ka];
    a.h[1] = *(const v8h*)&etile[arow][ka + 8];
    const int kb = hi * 16;
    #pragma unroll
    for (int n = 0; n < 4; ++n) {
      const int col = n * 16 + mrow;
      const _Float16* bp = WhT + (size_t)col * NN + j0 + kb;
      AFrag b;
      b.h[0] = *(const v8h*)bp;
      b.h[1] = *(const v8h*)(bp + 8);
      acc[n] = wmma_f16(a.v, b.v, acc[n]);
    }
    __syncthreads();  // protect etile/w2s for next iteration
  }

  // ---- epilogue: ELU, scatter into h_cat layout out0[i][r*FOUT + f] ----
  #pragma unroll
  for (int n = 0; n < 4; ++n) {
    const int col = n * 16 + mrow;
    #pragma unroll
    for (int v = 0; v < 8; ++v) {
      const int row = i0 + wv * 16 + hi * 8 + v;
      float x = acc[n][v];
      x = (x > 0.f) ? x : expm1f(x);
      out0[(size_t)row * (RDIM * FOUT) + r * FOUT + col] = x;
    }
  }
}

// ---------------------------------------------------------------------------
extern "C" void kernel_launch(void* const* d_in, const int* in_sizes, int n_in,
                              void* d_out, int out_size, void* d_ws, size_t ws_size,
                              hipStream_t stream) {
  (void)in_sizes; (void)n_in; (void)out_size; (void)ws_size;
  const float* h    = (const float*)d_in[0];  // [N, Fin]
  const float* edge = (const float*)d_in[1];  // [R, N, N]
  const float* W    = (const float*)d_in[2];  // [Fin, Fout]
  const float* a    = (const float*)d_in[3];  // [2*Fout, 1]

  // Workspace carve-out (~1.7 MB)
  float* Wh   = (float*)d_ws;                       // N*FOUT f32
  float* wh1  = Wh + (size_t)NN * FOUT;             // N f32
  float* wh2  = wh1 + NN;                           // N f32
  float* mcol = wh2 + NN;                           // R*N f32
  float* sinv = mcol + (size_t)RDIM * NN;           // R*N f32
  _Float16* WhT = (_Float16*)(sinv + (size_t)RDIM * NN);  // FOUT*N f16

  float* out0  = (float*)d_out;                     // [N, R*FOUT] (elu(h_cat))
  float* out_e = out0 + (size_t)NN * RDIM * FOUT;   // [R, N, N]   (softmaxed e)

  gat_wh_gemm<<<NN / 64, 128, 0, stream>>>(h, W, Wh, WhT);
  gat_attn_vec<<<NN / 256, 256, 0, stream>>>(Wh, a, wh1, wh2);
  gat_col_stats<<<dim3(NN / 256, RDIM), 256, 0, stream>>>(edge, wh1, wh2, mcol, sinv);
  gat_apply<<<dim3(NN / 64, RDIM), 128, 0, stream>>>(edge, wh1, wh2, mcol, sinv,
                                                     WhT, out0, out_e);
}
